// MultiHeadAttention_untie_39822936768826
// MI455X (gfx1250) — matrix operand order repaired and checked by reading, not compiled, not run on hardware
//
#include <hip/hip_runtime.h>
#include <hip/hip_bf16.h>
#include <math.h>

// ---------------------------------------------------------------------------
// MHA forward for MI455X (gfx1250, wave32, WMMA).
// Pipeline: fp32->bf16 cvt -> 3 WMMA projection GEMMs (V transposed per head)
//           -> flash-attention (WMMA QK^T, online softmax, WMMA P*V)
//           -> WMMA output projection (fp32 out).
// ---------------------------------------------------------------------------

typedef __attribute__((ext_vector_type(8)))  __bf16 v8bf;
typedef __attribute__((ext_vector_type(16))) __bf16 v16bf;
typedef __attribute__((ext_vector_type(8)))  float  v8f;

#define BB  4
#define SS  2048
#define DD  768
#define HH  12
#define DKK 64
#define MM  (BB * SS)

__device__ __forceinline__ unsigned short f2bf(float f) {
  unsigned int u = __float_as_uint(f);
  u += 0x7FFFu + ((u >> 16) & 1u);           // round-to-nearest-even
  return (unsigned short)(u >> 16);
}

__device__ __forceinline__ v16bf ld16(const unsigned short* p) {
  return *(const v16bf*)p;
}
__device__ __forceinline__ v8bf ld8(const unsigned short* p) {
  return *(const v8bf*)p;
}
__device__ __forceinline__ v16bf cat16(v8bf lo, v8bf hi) {
  return __builtin_shufflevector(lo, hi, 0,1,2,3,4,5,6,7,8,9,10,11,12,13,14,15);
}
__device__ __forceinline__ v8f wmma_bf16(v16bf a, v16bf b, v8f c) {
  // (neg_a, A, neg_b, B, c_mod, C, reuse_a, reuse_b)
  return __builtin_amdgcn_wmma_f32_16x16x32_bf16(false, a, false, b, (short)0, c,
                                                 false, false);
}

// ---------------------------------------------------------------------------
// fp32 -> bf16 elementwise conversion
// ---------------------------------------------------------------------------
__global__ __launch_bounds__(256) void cvt_bf16(const float* __restrict__ src,
                                                unsigned short* __restrict__ dst,
                                                int n) {
  int i = blockIdx.x * blockDim.x + threadIdx.x;
  int stride = gridDim.x * blockDim.x;
  for (; i < n; i += stride) dst[i] = f2bf(src[i]);
}

// ---------------------------------------------------------------------------
// WMMA GEMM: Y[m,n] = sum_k A[m,k] * W[n,k] + bias[n]
// M=8192, N=768, K=768. Block = 256 thr = 8 waves, each wave does 16x64.
// AMODE: 0 = A flat row-major [M,K]; 1 = A is head-major [b,h,s,dk] ("concat")
// OMODE: 0 = bf16 head-major [b,h,s,dk]; 1 = bf16 transposed [b,h,dk,s];
//        2 = fp32 flat [M,N]
// ---------------------------------------------------------------------------
template <int AMODE, int OMODE>
__global__ __launch_bounds__(256) void gemm_wmma(
    const unsigned short* __restrict__ A, const unsigned short* __restrict__ W,
    const float* __restrict__ bias, float* __restrict__ outf,
    unsigned short* __restrict__ outb) {
  const int lane = threadIdx.x & 31;
  const int wave = threadIdx.x >> 5;
  const int c16  = lane & 15;
  const int half = lane >> 4;
  const int ko   = half ? 16 : 0;    // A-frag K sub-offset per lane group
  const int kb   = half ? 16 : 0;    // B-frag K block per lane group
  const int row_base = blockIdx.x * 128 + wave * 16;
  const int col_base = blockIdx.y * 64;

  float bval[4];
#pragma unroll
  for (int c = 0; c < 4; ++c) bval[c] = bias[col_base + 16 * c + c16];

  v8f acc[4] = {};
  const int m = row_base + c16;  // A-frag row owned by this lane

  for (int k0 = 0; k0 < DD; k0 += 32) {
    v16bf afrag;
    if (AMODE == 0) {
      const unsigned short* ap = A + (size_t)m * DD + k0;
      afrag = cat16(ld8(ap + ko), ld8(ap + 8 + ko));
    } else {
      // A is attention output [b,h,s,dk]; flat col e -> (h=e/64, dk=e%64)
      const int bi = m / SS, s = m % SS;
      const int e0 = k0 + ko, e1 = k0 + 8 + ko;
      const unsigned short* p0 =
          A + (((size_t)bi * HH + (e0 >> 6)) * SS + s) * DKK + (e0 & 63);
      const unsigned short* p1 =
          A + (((size_t)bi * HH + (e1 >> 6)) * SS + s) * DKK + (e1 & 63);
      afrag = cat16(ld8(p0), ld8(p1));
    }
#pragma unroll
    for (int c = 0; c < 4; ++c) {
      const int n = col_base + 16 * c + c16;
      v16bf bfrag = ld16(W + (size_t)n * DD + k0 + kb);  // W row n, contiguous k
      acc[c] = wmma_bf16(afrag, bfrag, acc[c]);
    }
  }

#pragma unroll
  for (int c = 0; c < 4; ++c) {
    const int n = col_base + 16 * c + c16;
#pragma unroll
    for (int r = 0; r < 8; ++r) {
      const int rowg = row_base + (half ? 8 + r : r);  // C layout: row/col map
      const float val = acc[c][r] + bval[c];
      if (OMODE == 2) {
        outf[(size_t)rowg * DD + n] = val;
      } else {
        const int bi = rowg / SS, s = rowg % SS;
        const int h = n >> 6, dk = n & 63;
        size_t addr;
        if (OMODE == 0)
          addr = (((size_t)bi * HH + h) * SS + s) * DKK + dk;  // [b,h,s,dk]
        else
          addr = (((size_t)bi * HH + h) * DKK + dk) * SS + s;  // [b,h,dk,s]
        outb[addr] = f2bf(val);
      }
    }
  }
}

// ---------------------------------------------------------------------------
// Flash attention: each wave owns 16 query rows of one (b,h); streams 32-key
// tiles. Causal mask == the tril mask input. scale = 1/sqrt(2*DK).
// ---------------------------------------------------------------------------
__global__ __launch_bounds__(128) void attn_wmma(
    const unsigned short* __restrict__ Qh, const unsigned short* __restrict__ Kh,
    const unsigned short* __restrict__ VhT, const float* __restrict__ pe,
    unsigned short* __restrict__ outh) {
  __shared__ __align__(16) unsigned short Pl[4][16][32];  // per-wave P tile

  const int lane = threadIdx.x & 31;
  const int wave = threadIdx.x >> 5;
  const int c16  = lane & 15;
  const int half = lane >> 4;
  const int ko   = half ? 16 : 0;
  const int kb   = half ? 16 : 0;
  const int bh   = blockIdx.y;            // b*H + h
  const int h    = bh % HH;
  const int qbase = blockIdx.x * 64 + wave * 16;

  const float scale = 0.08838834764831845f;  // 1/sqrt(128)

  const unsigned short* Qb = Qh + (size_t)bh * SS * DKK;
  const unsigned short* Kb = Kh + (size_t)bh * SS * DKK;
  const unsigned short* Vb = VhT + (size_t)bh * DKK * SS;
  const float* peb = pe + (size_t)h * SS * SS;

  // Q A-fragments for DK chunks [0,32) and [32,64)
  const unsigned short* qp = Qb + (size_t)(qbase + c16) * DKK;
  const v16bf aq0 = cat16(ld8(qp + ko), ld8(qp + 8 + ko));
  const v16bf aq1 = cat16(ld8(qp + 32 + ko), ld8(qp + 40 + ko));

  float mrow[8], lrow[8];
  v8f o[4] = {};
#pragma unroll
  for (int r = 0; r < 8; ++r) { mrow[r] = -INFINITY; lrow[r] = 0.f; }

  const int jend = qbase + 16;  // last key tile containing unmasked entries
  for (int j0 = 0; j0 < jend; j0 += 32) {
    if (j0 + 32 < jend)  // prefetch next K tile (global_prefetch_b8)
      __builtin_prefetch(Kb + (size_t)(j0 + 32 + c16) * DKK, 0, 0);

    // S = Q * K^T : two 16-key column subtiles, two DK chunks each
    v8f sf[2] = {};
#pragma unroll
    for (int nh = 0; nh < 2; ++nh) {
      const unsigned short* kp = Kb + (size_t)(j0 + 16 * nh + c16) * DKK;
      sf[nh] = wmma_bf16(aq0, ld16(kp + kb), sf[nh]);
      sf[nh] = wmma_bf16(aq1, ld16(kp + 32 + kb), sf[nh]);
    }

    // scale + pe + causal mask, online softmax (rows live on 16-lane groups)
    float p0[8], p1[8], corr[8];
#pragma unroll
    for (int r = 0; r < 8; ++r) {
      const int q = qbase + (half ? 8 + r : r);
      const int col0 = j0 + c16, col1 = j0 + 16 + c16;
      float s0 = sf[0][r] * scale + peb[(size_t)q * SS + col0];
      float s1 = sf[1][r] * scale + peb[(size_t)q * SS + col1];
      if (col0 > q) s0 = -100000000.f;
      if (col1 > q) s1 = -100000000.f;
      float t = fmaxf(s0, s1);
#pragma unroll
      for (int mk = 8; mk >= 1; mk >>= 1) t = fmaxf(t, __shfl_xor(t, mk, 32));
      const float mnew = fmaxf(mrow[r], t);
      corr[r] = __expf(mrow[r] - mnew);
      p0[r] = __expf(s0 - mnew);
      p1[r] = __expf(s1 - mnew);
      float ps = p0[r] + p1[r];
#pragma unroll
      for (int mk = 8; mk >= 1; mk >>= 1) ps += __shfl_xor(ps, mk, 32);
      lrow[r] = lrow[r] * corr[r] + ps;
      mrow[r] = mnew;
    }

    // rescale running output accumulators
#pragma unroll
    for (int c = 0; c < 4; ++c)
#pragma unroll
      for (int r = 0; r < 8; ++r) o[c][r] *= corr[r];

    // C-layout P -> LDS (row-major 16x32 bf16) -> A-layout fragment
#pragma unroll
    for (int r = 0; r < 8; ++r) {
      const int row = half ? 8 + r : r;
      Pl[wave][row][c16]      = f2bf(p0[r]);
      Pl[wave][row][16 + c16] = f2bf(p1[r]);
    }
    asm volatile("s_wait_dscnt 0" ::: "memory");  // same-wave LDS RAW fence

    const v16bf pa =
        cat16(ld8(&Pl[wave][c16][ko]), ld8(&Pl[wave][c16][8 + ko]));

    // O += P * V ; V^T layout makes B-frags contiguous 16-elem loads
#pragma unroll
    for (int c = 0; c < 4; ++c) {
      const unsigned short* vp = Vb + (size_t)(16 * c + c16) * SS + j0 + kb;
      o[c] = wmma_bf16(pa, ld16(vp), o[c]);
    }
  }

  // normalize and store head-major bf16
#pragma unroll
  for (int c = 0; c < 4; ++c)
#pragma unroll
    for (int r = 0; r < 8; ++r) {
      const int q = qbase + (half ? 8 + r : r);
      outh[((size_t)bh * SS + q) * DKK + 16 * c + c16] = f2bf(o[c][r] / lrow[r]);
    }
}

// ---------------------------------------------------------------------------
extern "C" void kernel_launch(void* const* d_in, const int* in_sizes, int n_in,
                              void* d_out, int out_size, void* d_ws,
                              size_t ws_size, hipStream_t stream) {
  (void)in_sizes; (void)n_in; (void)out_size; (void)ws_size;
  const float* q  = (const float*)d_in[0];
  const float* k  = (const float*)d_in[1];
  const float* v  = (const float*)d_in[2];
  const float* pe = (const float*)d_in[3];
  // d_in[4] = mask == tril -> implemented as causal predicate in-kernel
  const float* Wq = (const float*)d_in[5];
  const float* bq = (const float*)d_in[6];
  const float* Wk = (const float*)d_in[7];
  const float* bk = (const float*)d_in[8];
  const float* Wv = (const float*)d_in[9];
  const float* bv = (const float*)d_in[10];
  const float* Wo = (const float*)d_in[11];
  const float* bo = (const float*)d_in[12];
  float* out = (float*)d_out;

  // workspace carve-up (all sizes 512B-multiples)
  char* ws = (char*)d_ws;
  const size_t SZ_X = (size_t)MM * DD * 2;  // 12.58 MB (bf16 [M,D] / [b,h,*,*])
  const size_t SZ_W = (size_t)DD * DD * 2;  // 1.18 MB
  unsigned short* q_bf  = (unsigned short*)ws; ws += SZ_X;
  unsigned short* k_bf  = (unsigned short*)ws; ws += SZ_X;
  unsigned short* v_bf  = (unsigned short*)ws; ws += SZ_X;
  unsigned short* Wq_bf = (unsigned short*)ws; ws += SZ_W;
  unsigned short* Wk_bf = (unsigned short*)ws; ws += SZ_W;
  unsigned short* Wv_bf = (unsigned short*)ws; ws += SZ_W;
  unsigned short* Wo_bf = (unsigned short*)ws; ws += SZ_W;
  unsigned short* Qh    = (unsigned short*)ws; ws += SZ_X;  // [b,h,s,dk]
  unsigned short* Kh    = (unsigned short*)ws; ws += SZ_X;  // [b,h,s,dk]
  unsigned short* VhT   = (unsigned short*)ws; ws += SZ_X;  // [b,h,dk,s]
  unsigned short* outh  = (unsigned short*)ws; ws += SZ_X;  // [b,h,s,dk]

  const int nX = MM * DD, nW = DD * DD;
  cvt_bf16<<<2048, 256, 0, stream>>>(q, q_bf, nX);
  cvt_bf16<<<2048, 256, 0, stream>>>(k, k_bf, nX);
  cvt_bf16<<<2048, 256, 0, stream>>>(v, v_bf, nX);
  cvt_bf16<<<512, 256, 0, stream>>>(Wq, Wq_bf, nW);
  cvt_bf16<<<512, 256, 0, stream>>>(Wk, Wk_bf, nW);
  cvt_bf16<<<512, 256, 0, stream>>>(Wv, Wv_bf, nW);
  cvt_bf16<<<512, 256, 0, stream>>>(Wo, Wo_bf, nW);

  dim3 gg(MM / 128, DD / 64);  // 64 x 12
  gemm_wmma<0, 0><<<gg, 256, 0, stream>>>(q_bf, Wq_bf, bq, nullptr, Qh);
  gemm_wmma<0, 0><<<gg, 256, 0, stream>>>(k_bf, Wk_bf, bk, nullptr, Kh);
  gemm_wmma<0, 1><<<gg, 256, 0, stream>>>(v_bf, Wv_bf, bv, nullptr, VhT);

  dim3 ga(SS / 64, BB * HH);  // 32 x 48
  attn_wmma<<<ga, 128, 0, stream>>>(Qh, Kh, VhT, pe, outh);

  gemm_wmma<1, 2><<<gg, 256, 0, stream>>>(outh, Wo_bf, bo, out, nullptr);
}
